// WindowAttentionV2_46849503264920
// MI455X (gfx1250) — matrix operand order repaired and checked by reading, not compiled, not run on hardware
//
#include <hip/hip_runtime.h>
#include <hip/hip_bf16.h>
#include <stdint.h>

typedef __attribute__((ext_vector_type(16))) __bf16 v16bf;
typedef __attribute__((ext_vector_type(8)))  float  v8f;

#define NTOK   64
#define DIMC   384
#define NHEAD  12
#define HD     32
#define NWIN   1024
#define QKVN   1152          // 3*DIMC
#define LDX    (DIMC + 8)    // 392  x / attn-out rows (mult. of 8 -> 16B aligned)
#define LDQK   (2*DIMC + 8)  // 776  q|k rows
#define LDV    (NTOK + 8)    // 72   vT rows (d-major)
#define LDK    (DIMC + 8)    // 392  staged weight panel (col-major)
#define LDS_S  (NTOK + 1)    // 65
#define LDP    (NTOK + 8)    // 72

// LDS carve (bytes)
#define XS_B    (NTOK * LDX * 2)              // 50176
#define QK_B    (NTOK * LDQK * 2)             // 99328
#define VT_B    (DIMC * LDV * 2)              // 55296
#define ST_B    (NTOK * LDS_S * 4)            // 16640
#define PT_B    (NTOK * LDP * 2)              // 9216
#define PANEL_B (32 * LDK * 2)                // 25088
#define SCR_B   (2 * PANEL_B)                 // 50176 >= ST_B+PT_B (25856)
#define SMEM_B  (XS_B + QK_B + VT_B + SCR_B)  // 254976 < 320K/WGP

// workspace carve (bytes)
#define WS_BIAS 0
#define WS_WQT  (196608)                       // bias_g = 12*64*64*4
#define WS_WPT  (196608 + QKVN * DIMC * 2)     // wq_t   = 1152*384*2

// ---- CDNA5 async copy: global -> LDS, 16 bytes, tracked by ASYNCcnt ----
__device__ inline void async_b128(void* lds_dst, const void* gsrc) {
  asm volatile("global_load_async_to_lds_b128 %0, %1, off"
               :: "v"((unsigned)(uintptr_t)lds_dst),
                  "v"((unsigned long long)(uintptr_t)gsrc)
               : "memory");
}
__device__ inline void wait_async0() {
  asm volatile("s_wait_asynccnt 0x0" ::: "memory");
}

// stage one 32-column weight panel (rows of W^T) x 384 K into LDS (async)
__device__ inline void stage_panel(__bf16* __restrict__ dst,
                                   const __bf16* __restrict__ wsrc, int tid) {
  for (int c = tid; c < 32 * 48; c += 256) {
    const int rr = c / 48, ch = (c % 48) * 8;
    async_b128(&dst[rr * LDK + ch], wsrc + rr * DIMC + ch);
  }
}

// ---- WMMA operand loaders (wave32, contiguous k order: k = 16*(lane/16) + s in A and B) ----
// A 16x32 bf16 from LDS, row-major, ld multiple of 8; k0 multiple of 32.
__device__ inline v16bf load_a_bf16(const __bf16* __restrict__ base, int ld,
                                    int m0, int k0, int lane) {
  const __bf16* p0 = base + (m0 + (lane & 15)) * ld + k0 + (lane >> 4) * 16;
  v16bf a;
  __builtin_memcpy(&a, p0, 32);
  return a;
}

// B 32x16 bf16 from LDS stored column-major: bp[col*ldk + k], 16 contiguous K.
__device__ inline v16bf load_b_lds(const __bf16* __restrict__ bp, int ldk,
                                   int n0, int k0, int lane) {
  const __bf16* p = bp + (n0 + (lane & 15)) * ldk + k0 + ((lane >> 4) << 4);
  v16bf b;
  __builtin_memcpy(&b, p, 32);
  return b;
}

// ---- Kernel 0: weight transpose + bf16 convert (tiny) ----
__global__ __launch_bounds__(256)
void prep_weights_kernel(const float* __restrict__ qkv_w,   // [384][1152]
                         const float* __restrict__ proj_w,  // [384][384]
                         __bf16* __restrict__ wq_t,         // [1152][384]
                         __bf16* __restrict__ wp_t)         // [384][384] transposed
{
  const int gid = blockIdx.x * blockDim.x + threadIdx.x;
  const int nq = QKVN * DIMC;
  if (gid < nq) {
    const int n = gid / DIMC, k = gid % DIMC;
    wq_t[gid] = (__bf16)qkv_w[k * QKVN + n];
  } else if (gid < nq + DIMC * DIMC) {
    const int i = gid - nq;
    const int n = i / DIMC, k = i % DIMC;
    wp_t[i] = (__bf16)proj_w[k * DIMC + n];
  }
}

// ---- Kernel 1: continuous-position-bias MLP + gather + sigmoid ----
__global__ __launch_bounds__(256)
void cpb_bias_kernel(const float* __restrict__ table,   // [64*64][2]
                     const int*   __restrict__ index,   // [64*64]
                     const float* __restrict__ w1,      // [2][512]
                     const float* __restrict__ b1,      // [512]
                     const float* __restrict__ w2,      // [512][12]
                     float* __restrict__ bias_g)        // [12][64*64]
{
  const int gid = blockIdx.x * blockDim.x + threadIdx.x;
  if (gid >= NTOK * NTOK) return;
  const int r  = index[gid];
  const float c0 = table[2 * r + 0];
  const float c1 = table[2 * r + 1];
  float acc[NHEAD];
#pragma unroll
  for (int h = 0; h < NHEAD; ++h) acc[h] = 0.f;
  for (int s = 0; s < 512; ++s) {
    const float hv = fmaxf(c0 * w1[s] + c1 * w1[512 + s] + b1[s], 0.f);
#pragma unroll
    for (int h = 0; h < NHEAD; ++h) acc[h] += hv * w2[s * NHEAD + h];
  }
#pragma unroll
  for (int h = 0; h < NHEAD; ++h)
    bias_g[h * (NTOK * NTOK) + gid] = 16.f / (1.f + __expf(-acc[h]));
}

// ---- Kernel 2: fused window attention, one workgroup (8 wave32) per window ----
__global__ __launch_bounds__(256)
void win_attn_kernel(const float* __restrict__ x,        // [4096][64][384]
                     const float* __restrict__ mask,     // [1024][64][64]
                     const __bf16* __restrict__ wq_t,    // [1152][384] bf16 W^T
                     const float* __restrict__ qkv_b,    // [1152]
                     const __bf16* __restrict__ wp_t,    // [384][384]  bf16 W^T
                     const float* __restrict__ proj_b,   // [384]
                     const float* __restrict__ logit_scale, // [12]
                     const float* __restrict__ bias_g,   // [12][64][64]
                     float* __restrict__ out)            // [4096][64][384]
{
  __shared__ __align__(16) unsigned char smem[SMEM_B];
  __bf16* xs  = (__bf16*)smem;                        // x tile / attn-out tile
  __bf16* qk  = (__bf16*)(smem + XS_B);               // q|k rows [64][776]
  __bf16* vts = (__bf16*)(smem + XS_B + QK_B);        // vT [384][72] d-major
  unsigned char* scr = smem + XS_B + QK_B + VT_B;     // phase-overlaid scratch
  __bf16* bs0 = (__bf16*)scr;                         // weight panel buffer 0
  __bf16* bs1 = (__bf16*)(scr + PANEL_B);             // weight panel buffer 1
  float*  s_t = (float*)scr;                          // scores   (attention)
  __bf16* p_t = (__bf16*)(scr + ST_B);                // probs    (attention)

  const int b    = blockIdx.x;
  const int tid  = threadIdx.x;
  const int lane = tid & 31;
  const int wv   = tid >> 5;            // wave id 0..7
  const int m0w  = (wv & 3) * 16;       // wave's M tile in GEMM panels
  const int n0w  = (wv >> 2) * 16;      // wave's N tile within 32-col panel

  // ---- stage x window into LDS as bf16; prefetch first weight panel ----
  stage_panel(bs0, wq_t, tid);
  const float* xw = x + (size_t)b * NTOK * DIMC;
  for (int i = tid; i < NTOK * DIMC; i += 256) {
    const int r = i / DIMC, c = i % DIMC;
    xs[r * LDX + c] = (__bf16)xw[i];
  }
  wait_async0();
  __syncthreads();

  // ---- QKV GEMM: [64,384] x [384,1152]; 36 double-buffered 32-col panels ----
  for (int pn = 0; pn < QKVN / 32; ++pn) {
    __bf16* cur = (pn & 1) ? bs1 : bs0;
    __bf16* nxt = (pn & 1) ? bs0 : bs1;
    if (pn + 1 < QKVN / 32)               // overlap DMA of next panel with WMMA
      stage_panel(nxt, wq_t + (size_t)(pn + 1) * 32 * DIMC, tid);

    v8f acc = {};
    for (int k0 = 0; k0 < DIMC; k0 += 32) {
      v16bf a  = load_a_bf16(xs, LDX, m0w, k0, lane);
      v16bf bb = load_b_lds(cur, LDK, n0w, k0, lane);
      acc = __builtin_amdgcn_wmma_f32_16x16x32_bf16(false, a, false, bb,
                                                    (short)0, acc, false, false);
    }
    const int rbase = m0w + ((lane >> 4) << 3);
    const int col   = pn * 32 + n0w + (lane & 15);
    const float bv  = qkv_b[col];
    if (pn < 24) {                        // q|k region: token-major
#pragma unroll
      for (int j = 0; j < 8; ++j)
        qk[(rbase + j) * LDQK + col] = (__bf16)(acc[j] + bv);
    } else {                              // v region: store transposed (d-major)
      const int d = col - 2 * DIMC;
#pragma unroll
      for (int j = 0; j < 8; ++j)
        vts[d * LDV + rbase + j] = (__bf16)(acc[j] + bv);
    }
    wait_async0();
    __syncthreads();
  }

  // ---- l2-normalize q,k per (token, head); fold cosine logit scale into q ----
  for (int t = tid; t < NTOK * NHEAD; t += 256) {
    const int tok = t / NHEAD, hh = t % NHEAD;
    const float sc = __expf(fminf(logit_scale[hh], 4.60517019f)); // ln(100)
    __bf16* qp = &qk[tok * LDQK + hh * HD];
    float ss = 0.f;
#pragma unroll
    for (int i = 0; i < HD; ++i) { const float v = (float)qp[i]; ss += v * v; }
    float inv = sc / fmaxf(sqrtf(ss), 1e-12f);
#pragma unroll
    for (int i = 0; i < HD; ++i) qp[i] = (__bf16)((float)qp[i] * inv);
    __bf16* kp = &qk[tok * LDQK + DIMC + hh * HD];
    ss = 0.f;
#pragma unroll
    for (int i = 0; i < HD; ++i) { const float v = (float)kp[i]; ss += v * v; }
    inv = 1.f / fmaxf(sqrtf(ss), 1e-12f);
#pragma unroll
    for (int i = 0; i < HD; ++i) kp[i] = (__bf16)((float)kp[i] * inv);
  }
  __syncthreads();

  const float* mw = mask + (size_t)(b % NWIN) * NTOK * NTOK;

  // ---- per-head: S = qk^T + bias + mask, softmax, O = P.V ----
  for (int hh = 0; hh < NHEAD; ++hh) {
    const float* bh = bias_g + (size_t)hh * NTOK * NTOK;

    // S: 4x4 tiles of 16x16, K = 32 (single WMMA each); 2 tiles per wave
    for (int t = wv; t < 16; t += 8) {
      const int m0 = (t >> 2) * 16;
      const int n0 = (t & 3) * 16;
      v16bf a = load_a_bf16(qk + hh * HD, LDQK, m0, 0, lane);
      v16bf bb;  // k^T: B[kk][n] = k[token n][d=kk] -> 16 contiguous bf16
      __builtin_memcpy(&bb,
          &qk[(n0 + (lane & 15)) * LDQK + DIMC + hh * HD + ((lane >> 4) << 4)], 32);
      v8f acc = {};
      acc = __builtin_amdgcn_wmma_f32_16x16x32_bf16(false, a, false, bb,
                                                    (short)0, acc, false, false);
      const int rbase = m0 + ((lane >> 4) << 3);
      const int col   = n0 + (lane & 15);
#pragma unroll
      for (int j = 0; j < 8; ++j) {
        const int r = rbase + j;
        s_t[r * LDS_S + col] = acc[j] + bh[r * NTOK + col] + mw[r * NTOK + col];
      }
    }
    __syncthreads();

    // softmax: 4 lanes per row, shuffle-reduce (all 256 threads active)
    {
      const int row = tid >> 2;
      const int q0  = (tid & 3) * 16;
      float* sr = s_t + row * LDS_S + q0;
      float mx = -1e30f;
#pragma unroll
      for (int j = 0; j < 16; ++j) mx = fmaxf(mx, sr[j]);
      mx = fmaxf(mx, __shfl_xor(mx, 1, 32));
      mx = fmaxf(mx, __shfl_xor(mx, 2, 32));
      float sum = 0.f;
#pragma unroll
      for (int j = 0; j < 16; ++j) {
        const float e = __expf(sr[j] - mx);
        sr[j] = e;
        sum += e;
      }
      sum += __shfl_xor(sum, 1, 32);
      sum += __shfl_xor(sum, 2, 32);
      const float inv = 1.f / sum;
      __bf16* pr = p_t + row * LDP + q0;
#pragma unroll
      for (int j = 0; j < 16; ++j) pr[j] = (__bf16)(sr[j] * inv);
    }
    __syncthreads();

    // O = P[64,64] . V_h[64,32]: 4x2 tiles, one per wave, K-steps of 32
    {
      const int m0 = (wv >> 1) * 16;
      const int n0 = (wv & 1) * 16;
      v8f acc = {};
      for (int ks = 0; ks < 2; ++ks) {
        v16bf a = load_a_bf16(p_t, LDP, m0, ks * 32, lane);
        v16bf bb;  // B[kk][d] = vT[d][kk] -> 16 contiguous bf16 (d-major)
        const int d   = hh * HD + n0 + (lane & 15);
        const int kk0 = ks * 32 + ((lane >> 4) << 4);
        __builtin_memcpy(&bb, &vts[d * LDV + kk0], 32);
        acc = __builtin_amdgcn_wmma_f32_16x16x32_bf16(false, a, false, bb,
                                                      (short)0, acc, false, false);
      }
      const int rbase = m0 + ((lane >> 4) << 3);
      const int col   = hh * HD + n0 + (lane & 15);
#pragma unroll
      for (int j = 0; j < 8; ++j)
        xs[(rbase + j) * LDX + col] = (__bf16)acc[j];  // xs reused as attn-out
    }
    __syncthreads();
  }

  // ---- output projection: [64,384] x [384,384]; 12 double-buffered panels ----
  float* ow = out + (size_t)b * NTOK * DIMC;
  stage_panel(bs0, wp_t, tid);
  wait_async0();
  __syncthreads();
  for (int pn = 0; pn < DIMC / 32; ++pn) {
    __bf16* cur = (pn & 1) ? bs1 : bs0;
    __bf16* nxt = (pn & 1) ? bs0 : bs1;
    if (pn + 1 < DIMC / 32)
      stage_panel(nxt, wp_t + (size_t)(pn + 1) * 32 * DIMC, tid);

    v8f acc = {};
    for (int k0 = 0; k0 < DIMC; k0 += 32) {
      v16bf a  = load_a_bf16(xs, LDX, m0w, k0, lane);
      v16bf bb = load_b_lds(cur, LDK, n0w, k0, lane);
      acc = __builtin_amdgcn_wmma_f32_16x16x32_bf16(false, a, false, bb,
                                                    (short)0, acc, false, false);
    }
    const int rbase = m0w + ((lane >> 4) << 3);
    const int col   = pn * 32 + n0w + (lane & 15);
    const float bv  = proj_b[col];
#pragma unroll
    for (int j = 0; j < 8; ++j)
      ow[(rbase + j) * DIMC + col] = acc[j] + bv;
    wait_async0();
    __syncthreads();
  }
}

extern "C" void kernel_launch(void* const* d_in, const int* in_sizes, int n_in,
                              void* d_out, int out_size, void* d_ws, size_t ws_size,
                              hipStream_t stream) {
  (void)in_sizes; (void)n_in; (void)out_size; (void)ws_size;
  const float* x           = (const float*)d_in[0];
  const float* mask        = (const float*)d_in[1];
  const float* qkv_w       = (const float*)d_in[2];
  const float* qkv_b       = (const float*)d_in[3];
  const float* proj_w      = (const float*)d_in[4];
  const float* proj_b      = (const float*)d_in[5];
  const float* logit_scale = (const float*)d_in[6];
  const float* cpb_w1      = (const float*)d_in[7];
  const float* cpb_b1      = (const float*)d_in[8];
  const float* cpb_w2      = (const float*)d_in[9];
  const float* rel_table   = (const float*)d_in[10];
  const int*   rel_index   = (const int*)d_in[11];
  float* out = (float*)d_out;

  float*  bias_g = (float*)((char*)d_ws + WS_BIAS);  // [12][64][64] fp32
  __bf16* wq_t   = (__bf16*)((char*)d_ws + WS_WQT);  // [1152][384] bf16
  __bf16* wp_t   = (__bf16*)((char*)d_ws + WS_WPT);  // [384][384]  bf16

  const int prep_elems = QKVN * DIMC + DIMC * DIMC;
  prep_weights_kernel<<<(prep_elems + 255) / 256, 256, 0, stream>>>(
      qkv_w, proj_w, wq_t, wp_t);

  cpb_bias_kernel<<<(NTOK * NTOK + 255) / 256, 256, 0, stream>>>(
      rel_table, rel_index, cpb_w1, cpb_b1, cpb_w2, bias_g);

  win_attn_kernel<<<4096, 256, 0, stream>>>(
      x, mask, wq_t, qkv_b, wp_t, proj_b, logit_scale, bias_g, out);
}
